// Masker_67156108640630
// MI455X (gfx1250) — compile-verified, matrix operand order checked by reading, not run on hardware
//
#include <hip/hip_runtime.h>
#include <hip/hip_bf16.h>
#include <stdint.h>

// ---- problem constants (match reference) ----
#define B_   64
#define T_   32
#define K_   4
#define D_   128
#define H_   8
#define NL_  6
#define DH_  256
#define M_   (B_*T_)     // 2048 tokens
#define KB_  (K_*B_)     // 256 rollout rows
#define G3_  (3*DH_)     // 768 GRU gate width

typedef __attribute__((ext_vector_type(16))) __bf16 v16bf;
typedef __attribute__((ext_vector_type(8)))  float  v8f;

union FragU { v16bf v; uint4 q[2]; };

// ---------------------------------------------------------------------------
// WMMA fragment loaders.  CDNA5 16-bit A-matrix 16x32 layout (ISA 7.12.2):
//   lanes 0-15  : M = lane,    K = {0..7} in VGPR0-3, {16..23} in VGPR4-7
//   lanes 16-31 : M = lane-16, K = {8..15},           {24..31}
// B (weights stored [N,K] row-major for x @ W^T) uses the same row-major
// pattern with N in place of M.
// ---------------------------------------------------------------------------
__device__ __forceinline__ v16bf frag_ld(const __hip_bfloat16* base, int ld,
                                         int row0, int k0, int lane) {
  FragU f;
  int r  = row0 + (lane & 15);
  int ko = (lane & 16) ? 8 : 0;
  const uint4* p = (const uint4*)(base + (size_t)r * ld + k0 + ko);
  f.q[0] = p[0];      // K ko..ko+7
  f.q[1] = p[2];      // K ko+16..ko+23  (+32 bytes)
  return f.v;
}

// A-fragment from an 8-row LDS tile (rows 8..15 of the 16-row tile are zero)
__device__ __forceinline__ v16bf frag_ld8(const __hip_bfloat16* base, int ld,
                                          int k0, int lane) {
  FragU f;
  int r  = lane & 15;
  int ko = (lane & 16) ? 8 : 0;
  if (r < 8) {
    const uint4* p = (const uint4*)(base + (size_t)r * ld + k0 + ko);
    f.q[0] = p[0];
    f.q[1] = p[2];
  } else {
    f.q[0] = make_uint4(0u,0u,0u,0u);
    f.q[1] = make_uint4(0u,0u,0u,0u);
  }
  return f.v;
}

__device__ __forceinline__ v8f wmma_bf16(v16bf a, v16bf b, v8f c) {
  return __builtin_amdgcn_wmma_f32_16x16x32_bf16(false, a, false, b,
                                                 (short)0, c, false, false);
}

__device__ __forceinline__ float sigm(float x) { return 1.f / (1.f + __expf(-x)); }

__device__ __forceinline__ float rng01(uint32_t x) {
  x ^= x >> 17; x *= 0xed5ad4bbu;
  x ^= x >> 11; x *= 0xac4c1b51u;
  x ^= x >> 15; x *= 0x31848babu;
  x ^= x >> 14;
  return (float)(x >> 8) * (1.0f / 16777216.0f);
}

// ---------------------------------------------------------------------------
// Register-blocked WMMA GEMM:  C[M,N] = A[M,K] * W[N,K]^T (+bias) (relu?)
// Each wave produces a 32x64 tile: 2 M-frags x 4 N-frags -> 8 accumulators.
// Per k-step: 6 fragment loads feed 8 WMMAs (0.75 loads/WMMA).
// Grids chosen so (M/32)*(N/64) is an exact multiple of 8 waves/block.
// flags: 1=relu  2=accumulate into C  4=also emit bf16 mirror to Cbf
// ---------------------------------------------------------------------------
__global__ void __launch_bounds__(256) k_wmma_gemm(
    const __hip_bfloat16* __restrict__ A, const __hip_bfloat16* __restrict__ W,
    const float* __restrict__ bias, float* C, __hip_bfloat16* Cbf,
    int Mr, int N, int Kd, int flags)
{
  int lane = threadIdx.x & 31;
  int wave = threadIdx.x >> 5;
  int ngrp = N >> 6;                          // 64-column groups
  int grp  = blockIdx.x * 8 + wave;
  if (grp >= (Mr >> 5) * ngrp) return;        // wave-uniform
  int tm = (grp / ngrp) << 5;                 // 32 rows
  int tn = (grp % ngrp) << 6;                 // 64 cols

  v8f acc[2][4] = {};
  for (int k0 = 0; k0 < Kd; k0 += 32) {
    v16bf a0 = frag_ld(A, Kd, tm,      k0, lane);
    v16bf a1 = frag_ld(A, Kd, tm + 16, k0, lane);
#pragma unroll
    for (int n = 0; n < 4; ++n) {
      v16bf b = frag_ld(W, Kd, tn + n * 16, k0, lane);
      acc[0][n] = wmma_bf16(a0, b, acc[0][n]);
      acc[1][n] = wmma_bf16(a1, b, acc[1][n]);
    }
  }
#pragma unroll
  for (int mi = 0; mi < 2; ++mi) {
    int rbase = tm + mi * 16 + ((lane & 16) ? 8 : 0);
#pragma unroll
    for (int n = 0; n < 4; ++n) {
      int col  = tn + n * 16 + (lane & 15);
      float bv = bias ? bias[col] : 0.f;
#pragma unroll
      for (int r = 0; r < 8; ++r) {
        size_t idx = (size_t)(rbase + r) * N + col;
        float v = acc[mi][n][r] + bv;
        if (flags & 2) v += C[idx];
        if (flags & 1) v = fmaxf(v, 0.f);
        C[idx] = v;
        if (flags & 4) Cbf[idx] = __float2bfloat16(v);
      }
    }
  }
}

// ---- f32 -> bf16 weight conversion ----
__global__ void k_f2bf(const float* __restrict__ a, __hip_bfloat16* __restrict__ o, int n) {
  int i = blockIdx.x * 256 + threadIdx.x;
  if (i < n) o[i] = __float2bfloat16(a[i]);
}

// ---- embedding: hybrid = tok[inp] + pos[t] + sty[label[b]]; also keep tok emb ----
__global__ void __launch_bounds__(128) k_embed(
    const int* __restrict__ inp, const int* __restrict__ label,
    const float* __restrict__ tok, const float* __restrict__ pos,
    const float* __restrict__ sty,
    float* __restrict__ x, __hip_bfloat16* __restrict__ xbf,
    float* __restrict__ te)
{
  int row = blockIdx.x;            // b*T + t
  int j   = threadIdx.x;           // 0..127
  int b = row / T_, t = row % T_;
  float e = tok[(size_t)inp[row] * D_ + j];
  te[(size_t)row * D_ + j] = e;
  float v = e + pos[t * D_ + j] + sty[label[b] * D_ + j];
  x[(size_t)row * D_ + j]   = v;
  xbf[(size_t)row * D_ + j] = __float2bfloat16(v);
}

// ---- fused residual + LayerNorm (writes f32 + bf16 mirror) ----
__global__ void __launch_bounds__(128) k_add_ln(
    const float* x, const float* dly,
    const float* __restrict__ g, const float* __restrict__ bta,
    float* out, __hip_bfloat16* outbf)
{
  int row = blockIdx.x, j = threadIdx.x;
  __shared__ float red[4], red2[4];
  float v = x[(size_t)row * D_ + j] + dly[(size_t)row * D_ + j];
  float s = v;
  for (int o = 16; o; o >>= 1) s += __shfl_xor(s, o);
  if ((j & 31) == 0) red[j >> 5] = s;
  __syncthreads();
  float mean = (red[0] + red[1] + red[2] + red[3]) * (1.f / 128.f);
  float d  = v - mean;
  float s2 = d * d;
  for (int o = 16; o; o >>= 1) s2 += __shfl_xor(s2, o);
  if ((j & 31) == 0) red2[j >> 5] = s2;
  __syncthreads();
  float var = (red2[0] + red2[1] + red2[2] + red2[3]) * (1.f / 128.f);
  float o = d * rsqrtf(var + 1e-5f) * g[j] + bta[j];
  out[(size_t)row * D_ + j]   = o;
  outbf[(size_t)row * D_ + j] = __float2bfloat16(o);
}

// ---------------------------------------------------------------------------
// Attention: one block per (b,h); 32x32 scores, dh=16.  Q/K/V staged to LDS
// via CDNA5 async global->LDS loads (ASYNCcnt), then VALU softmax.
// ---------------------------------------------------------------------------
__global__ void __launch_bounds__(256) k_attn(
    const float* __restrict__ qkv, float* __restrict__ atto,
    __hip_bfloat16* __restrict__ attobf)
{
  __shared__ __align__(16) float q[32][16], kk[32][16], vv[32][16];
  __shared__ float sc[32][32];
  int bh = blockIdx.x; int b = bh >> 3, h = bh & 7;
  int tid = threadIdx.x;

  if (tid < 128) {                 // 16 bytes per thread per matrix
    int row = tid >> 2, c4 = (tid & 3) * 4;
    const float* gq = qkv + (size_t)(b * T_ + row) * (3 * D_) + h * 16 + c4;
    uint32_t lq = (uint32_t)(uintptr_t)&q[row][c4];
    uint32_t lk = (uint32_t)(uintptr_t)&kk[row][c4];
    uint32_t lv = (uint32_t)(uintptr_t)&vv[row][c4];
    const float* gk = gq + D_;
    const float* gv = gq + 2 * D_;
    asm volatile("global_load_async_to_lds_b128 %0, %1, off" :: "v"(lq), "v"(gq) : "memory");
    asm volatile("global_load_async_to_lds_b128 %0, %1, off" :: "v"(lk), "v"(gk) : "memory");
    asm volatile("global_load_async_to_lds_b128 %0, %1, off" :: "v"(lv), "v"(gv) : "memory");
  }
  asm volatile("s_wait_asynccnt 0" ::: "memory");
  __syncthreads();

  for (int p = tid; p < 1024; p += 256) {        // scores
    int t = p >> 5, s = p & 31;
    float acc = 0.f;
#pragma unroll
    for (int d = 0; d < 16; ++d) acc += q[t][d] * kk[s][d];
    sc[t][s] = acc * 0.25f;                      // /sqrt(16)
  }
  __syncthreads();
  if (tid < 32) {                                // softmax rows
    int t = tid;
    float mx = -1e30f;
    for (int s = 0; s < 32; ++s) mx = fmaxf(mx, sc[t][s]);
    float sum = 0.f;
    for (int s = 0; s < 32; ++s) { float e = __expf(sc[t][s] - mx); sc[t][s] = e; sum += e; }
    float inv = 1.f / sum;
    for (int s = 0; s < 32; ++s) sc[t][s] *= inv;
  }
  __syncthreads();
  for (int p = tid; p < 512; p += 256) {         // attn @ V
    int t = p >> 4, d = p & 15;
    float acc = 0.f;
    for (int s = 0; s < 32; ++s) acc += sc[t][s] * vv[s][d];
    size_t idx = (size_t)(b * T_ + t) * D_ + h * 16 + d;
    atto[idx]   = acc;
    attobf[idx] = __float2bfloat16(acc);
  }
}

// ---- zero-init GRU state ----
__global__ void k_zero(float* h, __hip_bfloat16* hbf, __hip_bfloat16* ep) {
  int i = blockIdx.x * 256 + threadIdx.x;
  if (i < B_ * DH_) { h[i] = 0.f; hbf[i] = __float2bfloat16(0.f); }
  if (i < B_ * D_)  ep[i] = __float2bfloat16(0.f);
}

// ---------------------------------------------------------------------------
// Main-line GRU gates + decoder logits + sample (B rows). gi/gh come from the
// two WMMA GEMMs (biases already folded in).
// ---------------------------------------------------------------------------
__global__ void __launch_bounds__(256) k_gru_main(
    const float* __restrict__ gi, const float* __restrict__ gh,
    float* hq, __hip_bfloat16* hbf,
    const float* __restrict__ te, const float* __restrict__ ctx,
    const float* __restrict__ dw, const float* __restrict__ db,
    __hip_bfloat16* epbf, int* masks, float* probs, int t)
{
  int b = blockIdx.x, j = threadIdx.x;      // 256 threads
  __shared__ float hrow[DH_];
  __shared__ float p0s[8], p1s[8];

  size_t gb = (size_t)b * G3_;
  float r  = sigm(gi[gb + j]        + gh[gb + j]);
  float z  = sigm(gi[gb + 256 + j]  + gh[gb + 256 + j]);
  float n  = tanhf(gi[gb + 512 + j] + r * gh[gb + 512 + j]);
  float ho = hq[(size_t)b * DH_ + j];
  float hn = (1.f - z) * n + z * ho;
  hrow[j] = hn;
  hq[(size_t)b * DH_ + j]  = hn;
  hbf[(size_t)b * DH_ + j] = __float2bfloat16(hn);
  __syncthreads();

  const float* e = te  + (size_t)(b * T_ + t) * D_;
  const float* c = ctx + (size_t)(b * T_ + t) * D_;
  float fi  = (j < 128) ? e[j] : c[j - 128];   // feat[j]
  float fi2 = hrow[j];                         // feat[j+256]
  float p0 = fi * dw[j]       + fi2 * dw[256 + j];
  float p1 = fi * dw[512 + j] + fi2 * dw[768 + j];
  for (int o = 16; o; o >>= 1) { p0 += __shfl_xor(p0, o); p1 += __shfl_xor(p1, o); }
  if ((j & 31) == 0) { p0s[j >> 5] = p0; p1s[j >> 5] = p1; }
  __syncthreads();
  if (j == 0) {
    float l0 = db[0], l1 = db[1];
    for (int w = 0; w < 8; ++w) { l0 += p0s[w]; l1 += p1s[w]; }
    float m   = fmaxf(l0, l1);
    float lse = m + logf(__expf(l0 - m) + __expf(l1 - m));
    float u   = rng01((uint32_t)(t * 0x9E3779B1u + b * 0x85EBCA77u + 0x42u));
    int   a   = (u < __expf(l1 - lse)) ? 1 : 0;
    masks[b * T_ + t]  = a;
    probs[t * B_ + b]  = (a ? l1 : l0) - lse;
    p0s[0] = (float)a;
  }
  __syncthreads();
  int a = (int)p0s[0];
  if (j < 128) epbf[(size_t)b * D_ + j] = __float2bfloat16(a ? e[j] : 0.f);
}

// ---------------------------------------------------------------------------
// Fused persistent rollout: block owns 8 of the K*B=256 rows and runs steps
// s=t+1..T-1 entirely in LDS.  Each step: g = x@Wih^T (+bih) and h@Whh^T
// (+bhh) via bf16 WMMA.  The A fragment is hoisted: loaded once per k-step
// and swept across the wave's 6 N-tiles (6 accumulator pairs, ~96 VGPRs).
// LDS: 8*256*4 + 8*256*2 + 8*128*2 + 2*8*768*4 = 62 KB.
// ---------------------------------------------------------------------------
__global__ void __launch_bounds__(256) k_rollout(
    const float* __restrict__ hq, const __hip_bfloat16* __restrict__ epbf_g,
    const __hip_bfloat16* __restrict__ wih, const __hip_bfloat16* __restrict__ whh,
    const float* __restrict__ bih, const float* __restrict__ bhh,
    const float* __restrict__ te, const float* __restrict__ ctx,
    const float* __restrict__ dw, const float* __restrict__ db,
    const int* __restrict__ masks_main, int* __restrict__ masks_roll, int t)
{
  __shared__ __align__(16) float          hs [8][DH_];
  __shared__ __align__(16) __hip_bfloat16 hsb[8][DH_];
  __shared__ __align__(16) __hip_bfloat16 xsb[8][D_];
  __shared__ float g_i[8][G3_];
  __shared__ float g_h[8][G3_];

  int tid = threadIdx.x, lane = tid & 31, wave = tid >> 5;
  int r0 = blockIdx.x * 8;

  for (int e = tid; e < 8 * DH_; e += 256) {       // init h (tiled over k)
    int rr = e >> 8, j = e & 255;
    int b = (r0 + rr) % B_;
    float hv = hq[(size_t)b * DH_ + j];
    hs[rr][j] = hv; hsb[rr][j] = __float2bfloat16(hv);
  }
  for (int e = tid; e < 8 * D_; e += 256) {        // init emb_prev
    int rr = e >> 7, j = e & 127;
    int b = (r0 + rr) % B_;
    xsb[rr][j] = epbf_g[(size_t)b * D_ + j];
  }
  for (int e = tid; e < 8 * (t + 1); e += 256) {   // mask prefix
    int rr = e / (t + 1), s = e % (t + 1);
    int kb = r0 + rr, b = kb % B_;
    masks_roll[kb * T_ + s] = masks_main[b * T_ + s];
  }
  {                                                // warm weights into caches
    const char* p1 = (const char*)wih;             // 192 KB
    const char* p2 = (const char*)whh;             // 384 KB
    for (int off = tid * 128; off < G3_ * D_ * 2;  off += 256 * 128) __builtin_prefetch(p1 + off, 0, 0);
    for (int off = tid * 128; off < G3_ * DH_ * 2; off += 256 * 128) __builtin_prefetch(p2 + off, 0, 0);
  }
  __syncthreads();

  for (int s = t + 1; s < T_; ++s) {
    // --- WMMA GEMM: wave w owns N-tiles {w, w+8, ..., w+40}
    {
      v8f ai[6] = {}, ah[6] = {};
#pragma unroll
      for (int k0 = 0; k0 < D_; k0 += 32) {        // x @ Wih^T
        v16bf a = frag_ld8(&xsb[0][0], D_, k0, lane);
#pragma unroll
        for (int i = 0; i < 6; ++i) {
          v16bf w = frag_ld(wih, D_, (wave + 8 * i) << 4, k0, lane);
          ai[i] = wmma_bf16(a, w, ai[i]);
        }
      }
#pragma unroll
      for (int k0 = 0; k0 < DH_; k0 += 32) {       // h @ Whh^T
        v16bf a = frag_ld8(&hsb[0][0], DH_, k0, lane);
#pragma unroll
        for (int i = 0; i < 6; ++i) {
          v16bf w = frag_ld(whh, DH_, (wave + 8 * i) << 4, k0, lane);
          ah[i] = wmma_bf16(a, w, ah[i]);
        }
      }
      if (lane < 16) {                 // rows 0..7 live in lanes 0..15
#pragma unroll
        for (int i = 0; i < 6; ++i) {
          int col = ((wave + 8 * i) << 4) + lane;
          float bi = bih[col], bh = bhh[col];
#pragma unroll
          for (int r = 0; r < 8; ++r) { g_i[r][col] = ai[i][r] + bi; g_h[r][col] = ah[i][r] + bh; }
        }
      }
    }
    __syncthreads();

    // --- GRU gates
    for (int e = tid; e < 8 * DH_; e += 256) {
      int rr = e >> 8, j = e & 255;
      float rg = sigm(g_i[rr][j]       + g_h[rr][j]);
      float zg = sigm(g_i[rr][256 + j] + g_h[rr][256 + j]);
      float ng = tanhf(g_i[rr][512 + j] + rg * g_h[rr][512 + j]);
      float hn = (1.f - zg) * ng + zg * hs[rr][j];
      hs[rr][j] = hn; hsb[rr][j] = __float2bfloat16(hn);
    }
    __syncthreads();

    // --- logits + sample: wave w owns row w
    {
      int rr = wave;
      int kb = r0 + rr, b = kb % B_;
      const float* e_t = te  + (size_t)(b * T_ + s) * D_;
      const float* c_t = ctx + (size_t)(b * T_ + s) * D_;
      float p0 = 0.f, p1 = 0.f;
      for (int i = lane; i < 512; i += 32) {
        float f = (i < 128) ? e_t[i] : (i < 256 ? c_t[i - 128] : hs[rr][i - 256]);
        p0 += f * dw[i];
        p1 += f * dw[512 + i];
      }
      for (int o = 16; o; o >>= 1) { p0 += __shfl_xor(p0, o); p1 += __shfl_xor(p1, o); }
      int a = 0;
      if (lane == 0) {
        float l0 = p0 + db[0], l1 = p1 + db[1];
        float m  = fmaxf(l0, l1);
        float pr = __expf(l1 - m) / (__expf(l0 - m) + __expf(l1 - m));
        float u  = rng01((uint32_t)(0x27d4eb2du + t * 131071u + s * 8191u + kb * 2654435761u));
        a = (u < pr) ? 1 : 0;
        masks_roll[kb * T_ + s] = a;
      }
      a = __shfl(a, 0);
#pragma unroll
      for (int c = lane; c < D_; c += 32)
        xsb[rr][c] = __float2bfloat16(a ? e_t[c] : 0.f);
    }
    __syncthreads();
  }
}

// ---------------------------------------------------------------------------
// Rewards: per b, compute clf() on original and masked sequences, then
// reward[t,b] = 10 * r_sty * (r_cp - 0.5)
// ---------------------------------------------------------------------------
__global__ void __launch_bounds__(64) k_reward(
    const int* __restrict__ masks, int kk,
    const int* __restrict__ inp, const int* __restrict__ label,
    const int* __restrict__ pm,
    const float* __restrict__ clfE, const float* __restrict__ clfW,
    float* __restrict__ rewards, int t)
{
  int b = blockIdx.x, j = threadIdx.x;       // 64 threads
  __shared__ float red[64];
  __shared__ float accs[2];

  float po = 0.f;
  for (int s = 0; s < T_; ++s) po += clfE[(size_t)inp[b * T_ + s] * 64 + j];
  red[j] = po * clfW[j];
  __syncthreads();
  for (int o = 32; o; o >>= 1) { if (j < o) red[j] += red[j + o]; __syncthreads(); }
  float pred_ori = red[0] * (1.f / 32.f);
  __syncthreads();

  if (j == 0) { accs[0] = 0.f; accs[1] = 0.f; }
  __syncthreads();

  for (int kkk = 0; kkk < kk; ++kkk) {
    int row = kkk * B_ + b;
    float pt = 0.f;
    for (int s = 0; s < T_; ++s) {
      int m   = masks[row * T_ + s];
      int tok = m ? inp[b * T_ + s] : 0;
      pt += clfE[(size_t)tok * 64 + j];
    }
    red[j] = pt * clfW[j];
    __syncthreads();
    for (int o = 32; o; o >>= 1) { if (j < o) red[j] += red[j + o]; __syncthreads(); }
    if (j == 0) {
      float pred_tgt = red[0] * (1.f / 32.f);
      float cp = 0.f, psum = 0.f;
      for (int s = 0; s < T_; ++s) {
        float p = (float)pm[b * T_ + s];
        psum += p;
        cp   += p * (float)masks[row * T_ + s];
      }
      accs[0] += cp / psum;
      accs[1] += (pred_tgt - pred_ori);
    }
    __syncthreads();
  }
  if (j == 0) {
    float lab   = (float)label[b];
    float r_sty = (1.f - 2.f * lab) * accs[1] / (float)kk;
    float r_cp  = accs[0] / (float)kk - 0.5f;
    rewards[t * B_ + b] = 10.f * r_sty * r_cp;
  }
}

// ---------------------------------------------------------------------------
extern "C" void kernel_launch(void* const* d_in, const int* in_sizes, int n_in,
                              void* d_out, int out_size, void* d_ws, size_t ws_size,
                              hipStream_t stream) {
  (void)in_sizes; (void)n_in; (void)out_size; (void)ws_size;
  const int*   inp   = (const int*)  d_in[0];
  const int*   label = (const int*)  d_in[1];
  const int*   pm    = (const int*)  d_in[2];
  const float* tok   = (const float*)d_in[4];
  const float* sty   = (const float*)d_in[5];
  const float* pos   = (const float*)d_in[6];
  const float* wqkv  = (const float*)d_in[7];
  const float* bqkv  = (const float*)d_in[8];
  const float* wo    = (const float*)d_in[9];
  const float* bo    = (const float*)d_in[10];
  const float* w1    = (const float*)d_in[11];
  const float* b1    = (const float*)d_in[12];
  const float* w2    = (const float*)d_in[13];
  const float* b2    = (const float*)d_in[14];
  const float* ln1g  = (const float*)d_in[15];
  const float* ln1b  = (const float*)d_in[16];
  const float* ln2g  = (const float*)d_in[17];
  const float* ln2b  = (const float*)d_in[18];
  const float* wih   = (const float*)d_in[19];
  const float* whh   = (const float*)d_in[20];
  const float* bih   = (const float*)d_in[21];
  const float* bhh   = (const float*)d_in[22];
  const float* dw    = (const float*)d_in[23];
  const float* db    = (const float*)d_in[24];
  const float* clfE  = (const float*)d_in[25];
  const float* clfW  = (const float*)d_in[26];
  float* out = (float*)d_out;            // probs [T*B] ++ rewards [T*B]

  // --- workspace carve (bump allocator, 256B aligned; ~19 MB total) ---
  char* wp = (char*)d_ws;
  auto alloc = [&](size_t bytes) -> void* {
    void* r = wp; wp += (bytes + 255) & ~(size_t)255; return r;
  };
  const int nWQKV = NL_*3*D_*D_, nWO = NL_*D_*D_, nW1 = NL_*4*D_*D_, nW2 = NL_*D_*4*D_;
  const int nWIH = G3_*D_, nWHH = G3_*DH_;
  __hip_bfloat16* wqkv_b = (__hip_bfloat16*)alloc((size_t)nWQKV*2);
  __hip_bfloat16* wo_b   = (__hip_bfloat16*)alloc((size_t)nWO*2);
  __hip_bfloat16* w1_b   = (__hip_bfloat16*)alloc((size_t)nW1*2);
  __hip_bfloat16* w2_b   = (__hip_bfloat16*)alloc((size_t)nW2*2);
  __hip_bfloat16* wih_b  = (__hip_bfloat16*)alloc((size_t)nWIH*2);
  __hip_bfloat16* whh_b  = (__hip_bfloat16*)alloc((size_t)nWHH*2);
  float*          x      = (float*)alloc((size_t)M_*D_*4);
  __hip_bfloat16* xbf    = (__hip_bfloat16*)alloc((size_t)M_*D_*2);
  float*          te     = (float*)alloc((size_t)M_*D_*4);
  float*          qkv    = (float*)alloc((size_t)M_*3*D_*4);
  float*          atto   = (float*)alloc((size_t)M_*D_*4);
  __hip_bfloat16* attobf = (__hip_bfloat16*)alloc((size_t)M_*D_*2);
  float*          proj   = (float*)alloc((size_t)M_*D_*4);
  float*          ffh    = (float*)alloc((size_t)M_*4*D_*4);
  __hip_bfloat16* ffhbf  = (__hip_bfloat16*)alloc((size_t)M_*4*D_*2);
  float*          ffo    = (float*)alloc((size_t)M_*D_*4);
  float*          gi     = (float*)alloc((size_t)B_*G3_*4);
  float*          gh     = (float*)alloc((size_t)B_*G3_*4);
  float*          hq     = (float*)alloc((size_t)B_*DH_*4);
  __hip_bfloat16* hbf    = (__hip_bfloat16*)alloc((size_t)B_*DH_*2);
  __hip_bfloat16* epbf   = (__hip_bfloat16*)alloc((size_t)B_*D_*2);
  int*            mmain  = (int*)alloc((size_t)B_*T_*4);
  int*            mroll  = (int*)alloc((size_t)KB_*T_*4);

  // --- weights -> bf16 ---
  k_f2bf<<<(nWQKV+255)/256,256,0,stream>>>(wqkv, wqkv_b, nWQKV);
  k_f2bf<<<(nWO  +255)/256,256,0,stream>>>(wo,   wo_b,   nWO);
  k_f2bf<<<(nW1  +255)/256,256,0,stream>>>(w1,   w1_b,   nW1);
  k_f2bf<<<(nW2  +255)/256,256,0,stream>>>(w2,   w2_b,   nW2);
  k_f2bf<<<(nWIH +255)/256,256,0,stream>>>(wih,  wih_b,  nWIH);
  k_f2bf<<<(nWHH +255)/256,256,0,stream>>>(whh,  whh_b,  nWHH);

  // --- embeddings ---
  k_embed<<<M_,128,0,stream>>>(inp, label, tok, pos, sty, x, xbf, te);

  // --- transformer encoder (6 layers); wave macro-tile = 32x64 ---
  for (int i = 0; i < NL_; ++i) {
    const __hip_bfloat16* Wq = wqkv_b + (size_t)i*3*D_*D_;
    const __hip_bfloat16* Wp = wo_b   + (size_t)i*D_*D_;
    const __hip_bfloat16* Wf1= w1_b   + (size_t)i*4*D_*D_;
    const __hip_bfloat16* Wf2= w2_b   + (size_t)i*D_*4*D_;
    k_wmma_gemm<<<(M_/32)*(3*D_/64)/8,256,0,stream>>>(xbf, Wq, bqkv+i*3*D_, qkv, nullptr, M_, 3*D_, D_, 0);
    k_attn<<<B_*H_,256,0,stream>>>(qkv, atto, attobf);
    k_wmma_gemm<<<(M_/32)*(D_/64)/8,256,0,stream>>>(attobf, Wp, bo+i*D_, proj, nullptr, M_, D_, D_, 0);
    k_add_ln<<<M_,128,0,stream>>>(x, proj, ln1g+i*D_, ln1b+i*D_, x, xbf);
    k_wmma_gemm<<<(M_/32)*(4*D_/64)/8,256,0,stream>>>(xbf, Wf1, b1+i*4*D_, ffh, ffhbf, M_, 4*D_, D_, 1|4);
    k_wmma_gemm<<<(M_/32)*(D_/64)/8,256,0,stream>>>(ffhbf, Wf2, b2+i*D_, ffo, nullptr, M_, D_, 4*D_, 0);
    k_add_ln<<<M_,128,0,stream>>>(x, ffo, ln2g+i*D_, ln2b+i*D_, x, xbf);
  }
  // ctx = x (f32)

  // --- GRU state init ---
  k_zero<<<(B_*DH_+255)/256,256,0,stream>>>(hq, hbf, epbf);

  // --- sequential policy loop over t ---
  const int gruBlocks = (B_/32)*(G3_/64)/8;   // 3 blocks (24 wave macro-tiles)
  for (int t = 0; t < T_; ++t) {
    k_wmma_gemm<<<gruBlocks,256,0,stream>>>(epbf, wih_b, bih, gi, nullptr, B_, G3_, D_,  0);
    k_wmma_gemm<<<gruBlocks,256,0,stream>>>(hbf,  whh_b, bhh, gh, nullptr, B_, G3_, DH_, 0);
    k_gru_main<<<B_,256,0,stream>>>(gi, gh, hq, hbf, te, x, dw, db, epbf, mmain, out, t);
    if (t < T_ - 1) {
      k_rollout<<<KB_/8,256,0,stream>>>(hq, epbf, wih_b, whh_b, bih, bhh,
                                        te, x, dw, db, mmain, mroll, t);
      k_reward<<<B_,64,0,stream>>>(mroll, K_, inp, label, pm, clfE, clfW, out + T_*B_, t);
    } else {
      k_reward<<<B_,64,0,stream>>>(mmain, 1, inp, label, pm, clfE, clfW, out + T_*B_, t);
    }
  }
}
